// Transformer_87866440941707
// MI455X (gfx1250) — matrix operand order repaired and checked by reading, not compiled, not run on hardware
//
#include <hip/hip_runtime.h>
#include <hip/hip_bf16.h>
#include <stdint.h>

// ---------------------------------------------------------------------------
// CDNA5 / gfx1250 fused transformer block.
// Heavy math: v_wmma_f32_16x16x32_bf16 (wave32 WMMA, reuse-A hints).
// Data movement: GLOBAL_LOAD_ASYNC_TO_LDS_B128 (ASYNCcnt) with a 2-stage
// double-buffered pipeline (steady-state loop peeled, unconditional waits).
// ---------------------------------------------------------------------------

typedef __attribute__((ext_vector_type(16))) __bf16 v16bf;
typedef __attribute__((ext_vector_type(8)))  float  v8f;

#define TILE       128
#define KSTEP      32
#define LDS_STRIDE 40   // 32 + 8 ushort pad (80B rows, keeps 16B alignment, breaks bank conflicts)

#if defined(__has_builtin)
#  if __has_builtin(__builtin_amdgcn_global_load_async_to_lds_b128)
#    define HAVE_ASYNC_LDS 1
#  endif
#endif
#ifndef HAVE_ASYNC_LDS
#  define HAVE_ASYNC_LDS 0
#endif

#if HAVE_ASYNC_LDS
#  if __has_builtin(__builtin_amdgcn_s_wait_asynccnt)
#    define WAIT_ASYNC(n) __builtin_amdgcn_s_wait_asynccnt(n)
#  else
#    define WAIT_ASYNC(n) asm volatile("s_wait_asynccnt %0" :: "i"(n) : "memory")
#  endif
typedef int v4i_b128 __attribute__((vector_size(16)));
typedef __attribute__((address_space(1))) v4i_b128* gas_v4i;
typedef __attribute__((address_space(3))) v4i_b128* las_v4i;
__device__ __forceinline__ void async_b128(const unsigned short* g, unsigned short* l) {
    __builtin_amdgcn_global_load_async_to_lds_b128(
        (gas_v4i)const_cast<unsigned short*>(g), (las_v4i)l, 0, 0);
}
#endif

__device__ __forceinline__ unsigned short f32_to_bf16(float f) {
    union { float f; unsigned int u; } c; c.f = f;
    unsigned int u = c.u;
    u += 0x7FFFu + ((u >> 16) & 1u);     // round-to-nearest-even
    return (unsigned short)(u >> 16);
}

union Frag16 { v16bf v; uint4 q[2]; };

// GEMM mode bits
#define GM_OUT_F32  1
#define GM_BIAS     2
#define GM_RELU     4
#define GM_RESID    8
#define GM_SCORES   16
#define GM_CAUSALK  32

// C[M,N] = A[M,K](bf16,row-major) * W[N,K]^T(bf16,row-major) (+bias,+resid,relu,...)
// Block: 256 threads = 8 waves; tile 128x128; wave w owns rows [16w,16w+16) x all 128 cols.
template<int MODE>
__global__ __launch_bounds__(256)
void gemm_bf16_wmma(const unsigned short* __restrict__ A, long long lda, long long strideAz,
                    const unsigned short* __restrict__ W, long long ldw, long long strideWz,
                    void* __restrict__ Out, long long ldo, long long strideOz,
                    const float* __restrict__ bias,
                    const float* __restrict__ resid, long long ldr, long long strideRz,
                    int K, float scale)
{
    const int nbase = blockIdx.x * TILE;
    const int mbase = blockIdx.y * TILE;
    const int z     = blockIdx.z;

    // causal scores: skip tiles fully above the diagonal (never read by softmax)
    if ((MODE & GM_SCORES) && nbase > mbase + (TILE - 1)) return;

    A += (long long)z * strideAz;
    W += (long long)z * strideWz;

    // double-buffered staging tiles (40KB total)
    __shared__ __align__(16) unsigned short As[2][TILE * LDS_STRIDE];
    __shared__ __align__(16) unsigned short Ws[2][TILE * LDS_STRIDE];

    const int tid  = threadIdx.x;
    const int lane = tid & 31;
    const int wave = tid >> 5;
    const int hi   = (lane >> 4) & 1;
    const int l15  = lane & 15;

    int kLimit = K;
    if (MODE & GM_CAUSALK) { int lim = mbase + TILE; kLimit = lim < K ? lim : K; }
    const int nsteps = kLimit / KSTEP;

    v8f acc[8];
#pragma unroll
    for (int i = 0; i < 8; ++i) acc[i] = (v8f){0.f,0.f,0.f,0.f,0.f,0.f,0.f,0.f};

    // each thread stages 32B of A-tile and 32B of W-tile per K step
    const int ldRow  = tid >> 1;          // 0..127
    const int ldHalf = (tid & 1) * 16;    // ushort offset (0 or 16)
    const unsigned short* gArow = A + (long long)(mbase + ldRow) * lda + ldHalf;
    const unsigned short* gWrow = W + (long long)(nbase + ldRow) * ldw + ldHalf;

    // one K-step of WMMA math out of LDS buffer `b`
    auto compute_step = [&](int b) {
        // A fragment: 16x32 bf16, ISA 7.12.2 layout:
        //   lanes 0-15: VGPR0-3 = K0..7, VGPR4-7 = K16..23 ; lanes 16-31: K8..15 / K24..31
        Frag16 afrag;
        const uint4* arow = (const uint4*)&As[b][(wave * 16 + l15) * LDS_STRIDE];
        afrag.q[0] = arow[hi];
        afrag.q[1] = arow[2 + hi];

        Frag16 bfrag[8];
#pragma unroll
        for (int ct = 0; ct < 8; ++ct) {
            const uint4* wrow = (const uint4*)&Ws[b][(ct * 16 + l15) * LDS_STRIDE];
            bfrag[ct].q[0] = wrow[hi];
            bfrag[ct].q[1] = wrow[2 + hi];
        }
        // same A operand feeds all 8 WMMAs: set the VOP3P reuse-A hint on all but the last
#pragma unroll
        for (int ct = 0; ct < 7; ++ct)
            acc[ct] = __builtin_amdgcn_wmma_f32_16x16x32_bf16(
                false, afrag.v, false, bfrag[ct].v, (short)0, acc[ct], true /*reuse_a*/, false);
        acc[7] = __builtin_amdgcn_wmma_f32_16x16x32_bf16(
            false, afrag.v, false, bfrag[7].v, (short)0, acc[7], false, false);
    };

#if HAVE_ASYNC_LDS
    auto issue_stage = [&](int s, int b) {
        const int k0 = s * KSTEP;
        unsigned short* lA = &As[b][ldRow * LDS_STRIDE + ldHalf];
        unsigned short* lW = &Ws[b][ldRow * LDS_STRIDE + ldHalf];
        async_b128(gArow + k0,     lA);
        async_b128(gArow + k0 + 8, lA + 8);
        async_b128(gWrow + k0,     lW);
        async_b128(gWrow + k0 + 8, lW + 8);
    };
    issue_stage(0, 0);
    int buf = 0;
    for (int s = 0; s < nsteps - 1; ++s) {      // steady state: unconditional issue+wait
        issue_stage(s + 1, buf ^ 1);            // prefetch next slab while current finishes
        WAIT_ASYNC(4);                          // wait only for current stage's 4 loads
        __syncthreads();
        compute_step(buf);
        __syncthreads();                        // readers done before buffer is refilled
        buf ^= 1;
    }
    WAIT_ASYNC(0);                              // drain final stage
    __syncthreads();
    compute_step(buf);
#else
    for (int s = 0; s < nsteps; ++s) {
        const int k0 = s * KSTEP;
        const uint4* ga = (const uint4*)(gArow + k0);
        const uint4* gw = (const uint4*)(gWrow + k0);
        uint4 a0 = ga[0], a1 = ga[1];
        uint4 w0 = gw[0], w1 = gw[1];
        uint4* la = (uint4*)&As[0][ldRow * LDS_STRIDE + ldHalf];
        uint4* lw = (uint4*)&Ws[0][ldRow * LDS_STRIDE + ldHalf];
        la[0] = a0; la[1] = a1;
        lw[0] = w0; lw[1] = w1;
        if (s + 1 < nsteps) {
            __builtin_prefetch(gArow + k0 + KSTEP, 0, 1);
            __builtin_prefetch(gWrow + k0 + KSTEP, 0, 1);
        }
        __syncthreads();
        compute_step(0);
        __syncthreads();
    }
#endif

    // Epilogue. C/D layout: VGPR r -> M = r (lanes 0-15) / 8+r (lanes 16-31); N = lane&15.
#pragma unroll
    for (int ct = 0; ct < 8; ++ct) {
        const int gn = nbase + ct * 16 + l15;
#pragma unroll
        for (int r = 0; r < 8; ++r) {
            const int gm = mbase + wave * 16 + r + ((lane >> 4) << 3);
            float v = acc[ct][r];
            if (MODE & GM_SCORES) { v *= scale; if (gn > gm) v = -1e30f; }
            if (MODE & GM_BIAS)   v += bias[gn];
            if (MODE & GM_RESID)  v += resid[(long long)z * strideRz + (long long)gm * ldr + gn];
            if (MODE & GM_RELU)   v = v > 0.f ? v : 0.f;
            if (MODE & GM_OUT_F32)
                ((float*)Out)[(long long)z * strideOz + (long long)gm * ldo + gn] = v;
            else
                ((unsigned short*)Out)[(long long)z * strideOz + (long long)gm * ldo + gn] = f32_to_bf16(v);
        }
    }
}

// Row LayerNorm (E=2048) -> bf16
__global__ __launch_bounds__(256)
void layernorm_bf16(const float* __restrict__ x, const float* __restrict__ gamma,
                    const float* __restrict__ beta, unsigned short* __restrict__ out, int E)
{
    const long long row = blockIdx.x;
    const float* xr = x + row * E;
    __shared__ float red[256];
    float s = 0.f, s2 = 0.f;
    for (int c = threadIdx.x; c < E; c += 256) { float v = xr[c]; s += v; s2 += v * v; }
    red[threadIdx.x] = s; __syncthreads();
    for (int o = 128; o > 0; o >>= 1) { if (threadIdx.x < o) red[threadIdx.x] += red[threadIdx.x + o]; __syncthreads(); }
    const float mean = red[0] / E; __syncthreads();
    red[threadIdx.x] = s2; __syncthreads();
    for (int o = 128; o > 0; o >>= 1) { if (threadIdx.x < o) red[threadIdx.x] += red[threadIdx.x + o]; __syncthreads(); }
    const float var = red[0] / E - mean * mean;
    const float rs  = rsqrtf(var + 1e-5f);
    for (int c = threadIdx.x; c < E; c += 256) {
        float v = (xr[c] - mean) * rs * gamma[c] + beta[c];
        out[row * E + c] = f32_to_bf16(v);
    }
}

// causal row softmax: S(f32) -> P(bf16), zero-padded out to the 128-row tile edge
__global__ __launch_bounds__(256)
void softmax_causal(const float* __restrict__ S, unsigned short* __restrict__ P, int T)
{
    const long long row = blockIdx.x;           // b*T + t
    const int t   = (int)(row % T);
    const int len = t + 1;
    const int pad = ((t >> 7) + 1) << 7;        // matches GM_CAUSALK K-limit
    const float* sr = S + row * (long long)T;
    unsigned short* pr = P + row * (long long)T;
    __shared__ float red[256];
    float m = -1e30f;
    for (int c = threadIdx.x; c < len; c += 256) m = fmaxf(m, sr[c]);
    red[threadIdx.x] = m; __syncthreads();
    for (int o = 128; o > 0; o >>= 1) { if (threadIdx.x < o) red[threadIdx.x] = fmaxf(red[threadIdx.x], red[threadIdx.x + o]); __syncthreads(); }
    m = red[0]; __syncthreads();
    float sum = 0.f;
    for (int c = threadIdx.x; c < len; c += 256) sum += __expf(sr[c] - m);
    red[threadIdx.x] = sum; __syncthreads();
    for (int o = 128; o > 0; o >>= 1) { if (threadIdx.x < o) red[threadIdx.x] += red[threadIdx.x + o]; __syncthreads(); }
    const float inv = 1.f / red[0];
    for (int c = threadIdx.x; c < len; c += 256) pr[c] = f32_to_bf16(__expf(sr[c] - m) * inv);
    for (int c = len + threadIdx.x; c < pad; c += 256) pr[c] = 0;
}

// V[b,s,h] (strided inside packed qkv) -> vT[b,h,s], 32x32 LDS tiles
__global__ __launch_bounds__(256)
void transpose_v(const unsigned short* __restrict__ qkv, unsigned short* __restrict__ vT,
                 int T, int HS, long long ldq, long long voff)
{
    __shared__ unsigned short tile[32][33];
    const int b  = blockIdx.z;
    const int s0 = blockIdx.x * 32, h0 = blockIdx.y * 32;
    const unsigned short* src = qkv + (long long)b * T * ldq + voff;
    for (int i = threadIdx.y; i < 32; i += 8)
        tile[i][threadIdx.x] = src[(long long)(s0 + i) * ldq + h0 + threadIdx.x];
    __syncthreads();
    unsigned short* dst = vT + (long long)b * HS * T;
    for (int i = threadIdx.y; i < 32; i += 8)
        dst[(long long)(h0 + i) * T + s0 + threadIdx.x] = tile[threadIdx.x][i];
}

__global__ __launch_bounds__(256)
void cvt_f32_bf16(const float* __restrict__ in, unsigned short* __restrict__ out, long long n)
{
    long long i = (long long)blockIdx.x * 256 + threadIdx.x;
    const long long stride = (long long)gridDim.x * 256;
    for (; i < n; i += stride) out[i] = f32_to_bf16(in[i]);
}

// ---------------------------------------------------------------------------

extern "C" void kernel_launch(void* const* d_in, const int* in_sizes, int n_in,
                              void* d_out, int out_size, void* d_ws, size_t ws_size,
                              hipStream_t stream)
{
    (void)in_sizes; (void)n_in; (void)out_size; (void)ws_size;

    constexpr int   Bb = 8, T = 2048, E = 2048, HS = 2048, FF = 8192;
    constexpr long long M = (long long)Bb * T;     // 16384 rows

    const float* x      = (const float*)d_in[0];
    const float* gamma1 = (const float*)d_in[1];
    const float* beta1  = (const float*)d_in[2];
    const float* Wq     = (const float*)d_in[3];
    const float* Wk     = (const float*)d_in[4];
    const float* Wv     = (const float*)d_in[5];
    const float* Wo     = (const float*)d_in[6];
    const float* bo     = (const float*)d_in[7];
    const float* gamma2 = (const float*)d_in[8];
    const float* beta2  = (const float*)d_in[9];
    const float* W1     = (const float*)d_in[10];
    const float* b1     = (const float*)d_in[11];
    const float* W2     = (const float*)d_in[12];
    const float* b2     = (const float*)d_in[13];
    float* out          = (float*)d_out;

    // workspace layout (bytes)
    char* ws = (char*)d_ws;
    auto take = [&](long long bytes) { char* p = ws; ws += (bytes + 255) & ~255LL; return p; };
    unsigned short* wqkv = (unsigned short*)take(3LL * E * E * 2);   // [3*HS, E]
    unsigned short* wo   = (unsigned short*)take((long long)E * E * 2);
    unsigned short* w1   = (unsigned short*)take((long long)FF * E * 2);
    unsigned short* w2   = (unsigned short*)take((long long)E * FF * 2);
    unsigned short* h1   = (unsigned short*)take(M * E * 2);
    unsigned short* qkv  = (unsigned short*)take(M * 3 * E * 2);     // [M, 6144]
    unsigned short* vT   = (unsigned short*)take((long long)Bb * E * T * 2);
    float*          Smat = (float*)take((long long)Bb * T * T * 4);
    unsigned short* P    = (unsigned short*)take((long long)Bb * T * T * 2);
    unsigned short* attn = (unsigned short*)take(M * E * 2);
    float*          x1   = (float*)take(M * E * 4);
    unsigned short* h2   = (unsigned short*)take(M * E * 2);
    unsigned short* ff   = (unsigned short*)take(M * FF * 2);

    // 1) weights -> bf16 (Wq/Wk/Wv packed into one [6144,2048])
    cvt_f32_bf16<<<4096, 256, 0, stream>>>(Wq, wqkv,                        (long long)E * E);
    cvt_f32_bf16<<<4096, 256, 0, stream>>>(Wk, wqkv + (long long)E * E,     (long long)E * E);
    cvt_f32_bf16<<<4096, 256, 0, stream>>>(Wv, wqkv + 2LL * E * E,          (long long)E * E);
    cvt_f32_bf16<<<4096, 256, 0, stream>>>(Wo, wo,                          (long long)E * E);
    cvt_f32_bf16<<<4096, 256, 0, stream>>>(W1, w1,                          (long long)FF * E);
    cvt_f32_bf16<<<4096, 256, 0, stream>>>(W2, w2,                          (long long)E * FF);

    // 2) LN1
    layernorm_bf16<<<(int)M, 256, 0, stream>>>(x, gamma1, beta1, h1, E);

    // 3) fused QKV projection: [M,6144] = h1 x wqkv^T
    gemm_bf16_wmma<0><<<dim3(3 * E / TILE, M / TILE, 1), 256, 0, stream>>>(
        h1, E, 0, wqkv, E, 0, qkv, 3 * E, 0, nullptr, nullptr, 0, 0, E, 0.f);

    // 4) V -> vT (per batch), V lives at column offset 2*HS inside qkv
    transpose_v<<<dim3(T / 32, HS / 32, Bb), dim3(32, 8), 0, stream>>>(
        qkv, vT, T, HS, 3 * E, 2LL * HS);

    // 5) causal scores: S = (Q K^T) / sqrt(E), upper tiles skipped, diag masked
    gemm_bf16_wmma<GM_SCORES | GM_OUT_F32><<<dim3(T / TILE, T / TILE, Bb), 256, 0, stream>>>(
        qkv /*Q*/, 3 * E, (long long)T * 3 * E,
        qkv + HS /*K*/, 3 * E, (long long)T * 3 * E,
        Smat, T, (long long)T * T, nullptr, nullptr, 0, 0, E, 0.022097087f /*1/sqrt(2048)*/);

    // 6) softmax rows -> P (bf16, zero-padded to tile edge)
    softmax_causal<<<(int)M, 256, 0, stream>>>(Smat, P, T);

    // 7) attn = P x V  (W = vT so it is A x W^T; causal K-limit)
    gemm_bf16_wmma<GM_CAUSALK><<<dim3(HS / TILE, T / TILE, Bb), 256, 0, stream>>>(
        P, T, (long long)T * T, vT, T, (long long)HS * T,
        attn, HS, (long long)T * HS, nullptr, nullptr, 0, 0, T, 0.f);

    // 8) x1 = x + attn x Wo^T + bo
    gemm_bf16_wmma<GM_OUT_F32 | GM_BIAS | GM_RESID><<<dim3(E / TILE, M / TILE, 1), 256, 0, stream>>>(
        attn, HS, 0, wo, HS, 0, x1, E, 0, bo, x, E, 0, HS, 0.f);

    // 9) LN2
    layernorm_bf16<<<(int)M, 256, 0, stream>>>(x1, gamma2, beta2, h2, E);

    // 10) ff = relu(h2 x W1^T + b1)
    gemm_bf16_wmma<GM_BIAS | GM_RELU><<<dim3(FF / TILE, M / TILE, 1), 256, 0, stream>>>(
        h2, E, 0, w1, E, 0, ff, FF, 0, b1, nullptr, 0, 0, E, 0.f);

    // 11) out = x1 + ff x W2^T + b2
    gemm_bf16_wmma<GM_OUT_F32 | GM_BIAS | GM_RESID><<<dim3(E / TILE, M / TILE, 1), 256, 0, stream>>>(
        ff, FF, 0, w2, FF, 0, out, E, 0, b2, x1, E, 0, FF, 0.f);
}
